// DiffEqSolver_29875792511583
// MI455X (gfx1250) — matrix operand order repaired and checked by reading, not compiled, not run on hardware
//
#include <hip/hip_runtime.h>

// Neural-ODE RK4 for MI455X (gfx1250): bf16 WMMA 16x16x32, f32 accumulate,
// persistent per-wave RK4 state, LDS-staged activations, L2-resident
// pre-swizzled bf16 weights.

typedef __attribute__((ext_vector_type(16))) __bf16 v16bf;
typedef __attribute__((ext_vector_type(8)))  float  v8f;

#define Bn 1024
#define Dn 512
#define Hn 1024
#define Tn 64

union ABFrag { uint4 q[2]; v16bf v; };

// round-to-nearest-even f32 -> bf16 (as u16 bits)
static __device__ __forceinline__ unsigned int f2bf(float f) {
    unsigned int u = __float_as_uint(f);
    u += 0x7FFFu + ((u >> 16) & 1u);
    return u >> 16;
}

// ---------------------------------------------------------------------------
// Pre-swizzle weights into WMMA B-fragment order (bf16).
// For each (kt,nt) 32x16 block: 32 lanes x 8 packed bf16-pairs, contiguous 1KB.
// Lane l: n = nt*16 + (l&15); pair i holds K rows {kt*32 + 2i + (l>>4)*16, +1}.
// ---------------------------------------------------------------------------
__global__ __launch_bounds__(256) void swizzle_weights(
    const float* __restrict__ W1, const float* __restrict__ W2,
    unsigned int* __restrict__ ws) {
    const int NU1 = (Dn * Hn) / 2;   // uints for W1 (512x1024)
    const int NU2 = (Hn * Dn) / 2;   // uints for W2 (1024x512)
    int u = blockIdx.x * 256 + threadIdx.x;
    const float* W;
    int Ncols;
    unsigned int* o;
    if (u < NU1) { W = W1; Ncols = Hn; o = ws; }
    else {
        u -= NU1;
        if (u >= NU2) return;
        W = W2; Ncols = Dn; o = ws + NU1;
    }
    const int i     = u & 7;
    const int lane  = (u >> 3) & 31;
    const int block = u >> 8;
    const int nt = block % (Ncols / 16);
    const int kt = block / (Ncols / 16);
    const int n  = nt * 16 + (lane & 15);
    const int k  = kt * 32 + 2 * i + (lane >> 4) * 16;
    const unsigned int lo = f2bf(W[(size_t)k * Ncols + n]);
    const unsigned int hi = f2bf(W[(size_t)(k + 1) * Ncols + n]);
    o[u] = lo | (hi << 16);
}

// ---------------------------------------------------------------------------
// Main persistent RK4 kernel: one block = 16 batch rows, 8 waves (wave32).
// Wave w owns output columns [w*64, w*64+64) of D (4 N-tiles) for GEMM2 and
// columns [w*128, w*128+128) of H (8 N-tiles) for GEMM1.
// ---------------------------------------------------------------------------
__global__ __launch_bounds__(256) void node_rk4(
    const float* __restrict__ z0, const float* __restrict__ tgrid,
    const unsigned int* __restrict__ W1sw, const float* __restrict__ b1,
    const unsigned int* __restrict__ W2sw, const float* __restrict__ b2,
    float* __restrict__ out) {
    __shared__ unsigned short zin[16 * Dn];    // bf16 A-operand for GEMM1 (16KB)
    __shared__ unsigned short hbuf[16 * Hn];   // bf16 A-operand for GEMM2 (32KB)

    const int tid   = threadIdx.x;
    const int wave  = tid >> 5;
    const int lane  = tid & 31;
    const int m0    = lane & 15;   // A-row / CD-column within tile
    const int khalf = lane >> 4;   // K-half for A, row-half for C/D
    const int row0  = blockIdx.x * 16;

    float zreg[4][8];   // persistent z state for wave-owned columns
    float sreg[4][8];   // RK4 running sum k1 + 2k2 + 2k3 (+ k4)
    float b1r[8], b2r[4];

#pragma unroll
    for (int jj = 0; jj < 8; ++jj) b1r[jj] = b1[(wave * 8 + jj) * 16 + m0];
#pragma unroll
    for (int jj = 0; jj < 4; ++jj) b2r[jj] = b2[(wave * 4 + jj) * 16 + m0];

    // Load z0 tile, emit traj[0], seed zin (bf16).
#pragma unroll
    for (int jj = 0; jj < 4; ++jj) {
        const int ncol = (wave * 4 + jj) * 16 + m0;
#pragma unroll
        for (int v = 0; v < 8; ++v) {
            const int m = khalf * 8 + v;
            const float z = z0[(size_t)(row0 + m) * Dn + ncol];
            zreg[jj][v] = z;
            out[(size_t)(row0 + m) * Dn + ncol] = z;
            zin[m * Dn + ncol] = (unsigned short)f2bf(z);
        }
    }

    // f(zin) = tanh(zin @ W1 + b1) @ W2 + b2, result in kout (4 N-tiles).
    auto feval = [&](v8f* kout) {
        __syncthreads();   // zin ready from all waves (also fences hbuf WAR)

        // ---- GEMM1: [16 x 512] x [512 x 1024], wave owns 8 N-tiles ----
        v8f acc1[8];
#pragma unroll
        for (int jj = 0; jj < 8; ++jj) {
            v8f a;
#pragma unroll
            for (int v = 0; v < 8; ++v) a[v] = b1r[jj];
            acc1[jj] = a;
        }
        for (int kc = 0; kc < Dn / 32; ++kc) {
            ABFrag af;
            const unsigned short* ap = &zin[m0 * Dn + kc * 32 + khalf * 16];
            af.q[0] = *(const uint4*)ap;
            af.q[1] = *(const uint4*)(ap + 8);
            const unsigned int* wb =
                W1sw + ((size_t)kc * (Hn / 16) + wave * 8) * 256 + lane * 8;
#pragma unroll
            for (int jj = 0; jj < 8; ++jj) {
                ABFrag bf;
                bf.q[0] = *(const uint4*)(wb);
                bf.q[1] = *(const uint4*)(wb + 4);
                wb += 256;
                acc1[jj] = __builtin_amdgcn_wmma_f32_16x16x32_bf16(
                    false, af.v, false, bf.v, (short)0, acc1[jj], false, false);
            }
        }
        // tanh + bf16, store activations to LDS
#pragma unroll
        for (int jj = 0; jj < 8; ++jj) {
            const int ncol = (wave * 8 + jj) * 16 + m0;
#pragma unroll
            for (int v = 0; v < 8; ++v) {
                hbuf[(khalf * 8 + v) * Hn + ncol] =
                    (unsigned short)f2bf(tanhf(acc1[jj][v]));
            }
        }
        __syncthreads();   // hbuf ready from all waves

        // ---- GEMM2: [16 x 1024] x [1024 x 512], wave owns 4 N-tiles ----
        v8f acc2[4];
#pragma unroll
        for (int jj = 0; jj < 4; ++jj) {
            v8f a;
#pragma unroll
            for (int v = 0; v < 8; ++v) a[v] = b2r[jj];
            acc2[jj] = a;
        }
        for (int kc = 0; kc < Hn / 32; ++kc) {
            ABFrag af;
            const unsigned short* ap = &hbuf[m0 * Hn + kc * 32 + khalf * 16];
            af.q[0] = *(const uint4*)ap;
            af.q[1] = *(const uint4*)(ap + 8);
            const unsigned int* wb =
                W2sw + ((size_t)kc * (Dn / 16) + wave * 4) * 256 + lane * 8;
#pragma unroll
            for (int jj = 0; jj < 4; ++jj) {
                ABFrag bf;
                bf.q[0] = *(const uint4*)(wb);
                bf.q[1] = *(const uint4*)(wb + 4);
                wb += 256;
                acc2[jj] = __builtin_amdgcn_wmma_f32_16x16x32_bf16(
                    false, af.v, false, bf.v, (short)0, acc2[jj], false, false);
            }
        }
#pragma unroll
        for (int jj = 0; jj < 4; ++jj) kout[jj] = acc2[jj];
    };

    for (int st = 0; st < Tn - 1; ++st) {
        const float dt = tgrid[st + 1] - tgrid[st];
        v8f kk[4];

        // ---- stage 1: k1 ----
        feval(kk);
#pragma unroll
        for (int jj = 0; jj < 4; ++jj) {
            const int ncol = (wave * 4 + jj) * 16 + m0;
#pragma unroll
            for (int v = 0; v < 8; ++v) {
                const float k = kk[jj][v];
                sreg[jj][v] = k;
                zin[(khalf * 8 + v) * Dn + ncol] =
                    (unsigned short)f2bf(zreg[jj][v] + 0.5f * dt * k);
            }
        }
        // ---- stage 2: k2 ----
        feval(kk);
#pragma unroll
        for (int jj = 0; jj < 4; ++jj) {
            const int ncol = (wave * 4 + jj) * 16 + m0;
#pragma unroll
            for (int v = 0; v < 8; ++v) {
                const float k = kk[jj][v];
                sreg[jj][v] += 2.0f * k;
                zin[(khalf * 8 + v) * Dn + ncol] =
                    (unsigned short)f2bf(zreg[jj][v] + 0.5f * dt * k);
            }
        }
        // ---- stage 3: k3 ----
        feval(kk);
#pragma unroll
        for (int jj = 0; jj < 4; ++jj) {
            const int ncol = (wave * 4 + jj) * 16 + m0;
#pragma unroll
            for (int v = 0; v < 8; ++v) {
                const float k = kk[jj][v];
                sreg[jj][v] += 2.0f * k;
                zin[(khalf * 8 + v) * Dn + ncol] =
                    (unsigned short)f2bf(zreg[jj][v] + dt * k);
            }
        }
        // ---- stage 4: k4 + combine + emit traj[st+1] ----
        feval(kk);
        float* op = out + (size_t)(st + 1) * Bn * Dn;
#pragma unroll
        for (int jj = 0; jj < 4; ++jj) {
            const int ncol = (wave * 4 + jj) * 16 + m0;
#pragma unroll
            for (int v = 0; v < 8; ++v) {
                const int m = khalf * 8 + v;
                const float znew =
                    zreg[jj][v] + (dt * (1.0f / 6.0f)) * (sreg[jj][v] + kk[jj][v]);
                op[(size_t)(row0 + m) * Dn + ncol] = znew;
                zreg[jj][v] = znew;
                zin[m * Dn + ncol] = (unsigned short)f2bf(znew);
            }
        }
    }
}

extern "C" void kernel_launch(void* const* d_in, const int* in_sizes, int n_in,
                              void* d_out, int out_size, void* d_ws, size_t ws_size,
                              hipStream_t stream) {
    const float* z0 = (const float*)d_in[0];
    const float* t  = (const float*)d_in[1];
    const float* W1 = (const float*)d_in[2];
    const float* b1 = (const float*)d_in[3];
    const float* W2 = (const float*)d_in[4];
    const float* b2 = (const float*)d_in[5];
    unsigned int* ws = (unsigned int*)d_ws;

    const int NU1 = (Dn * Hn) / 2;
    const int NU2 = (Hn * Dn) / 2;
    const int nThreads = NU1 + NU2;

    swizzle_weights<<<(nThreads + 255) / 256, 256, 0, stream>>>(W1, W2, ws);
    node_rk4<<<Bn / 16, 256, 0, stream>>>(z0, t, ws, b1, ws + NU1, b2,
                                          (float*)d_out);
}
